// DynamicsTransformer_35983236006581
// MI455X (gfx1250) — matrix operand order repaired and checked by reading, not compiled, not run on hardware
//
#include <hip/hip_runtime.h>
#include <hip/hip_bf16.h>
#include <math.h>

// ---------------------------------------------------------------- constants
#define B_   8
#define S_   2048
#define K_   128
#define D_   512
#define H_   8
#define DH_  64
#define LH_  4
#define W_   128
#define NW_  16
#define L_   4
#define BS_  (B_ * S_)          // 16384 rows
#define FD_  (4 * D_)           // 2048

typedef __attribute__((ext_vector_type(16))) __bf16 v16bf;
typedef __attribute__((ext_vector_type(2)))  __bf16 v2bf;
typedef __attribute__((ext_vector_type(8)))  float  v8f;

// ---------------------------------------------------------------- helpers
__device__ __forceinline__ unsigned pk2(float lo, float hi) {
  // native fptrunc -> v_cvt_pk_bf16_f32 on gfx1250
  v2bf v;
  v.x = (__bf16)lo;
  v.y = (__bf16)hi;
  return __builtin_bit_cast(unsigned, v);
}
__device__ __forceinline__ float gelu_f(float x) {
  return 0.5f * x * (1.f + erff(x * 0.70710678118654752f));
}
__device__ __forceinline__ v8f wmma_bf16(v16bf a, v16bf b, v8f c) {
  // D(f32 16x16) = A(16x32 bf16) * B(32x16 bf16) + C
  return __builtin_amdgcn_wmma_f32_16x16x32_bf16(false, a, false, b,
                                                 (short)0, c, false, false);
}

// ================================================================ weight prepack
// Convert W[K,N] (f32 row-major) into WMMA B-fragment order, bf16:
// slot (nt, kt, lane) holds 16 bf16: element i <-> K = kt*32 + 8*(lane>>4) + (i&7) + 16*(i>>3),
// col = nt*16 + (lane&15).  One slot (32 B) per thread.
__global__ __launch_bounds__(256)
void prep_w_k(const float* __restrict__ Wt, int ldw /* = N */, int Kdim,
              unsigned short* __restrict__ Wp) {
  const int gid  = blockIdx.x * 256 + threadIdx.x;
  const int lane = gid & 31;
  const int ktn  = gid >> 5;
  const int nkt  = Kdim >> 5;
  const int kt   = ktn % nkt;
  const int nt   = ktn / nkt;
  const int col  = nt * 16 + (lane & 15);
  const int krow = (kt << 5) + ((lane >> 4) << 3);
  const float* src = Wt + (size_t)krow * ldw + col;
  unsigned u[8];
  #pragma unroll
  for (int i = 0; i < 8; ++i)
    u[i] = pk2(src[(size_t)(2 * i + (i >> 2) * 8) * ldw], src[(size_t)(2 * i + 1 + (i >> 2) * 8) * ldw]);
  uint4* dst = (uint4*)(Wp + (size_t)gid * 16);
  dst[0] = make_uint4(u[0], u[1], u[2], u[3]);
  dst[1] = make_uint4(u[4], u[5], u[6], u[7]);
}

// ================================================================ GEMM
// C[M,N] = A[M,K] * W[K,N] (+epilogue).  Workgroup tile 128x64, 8 waves,
// each wave 32x32 (4 WMMA C frags).  A: double-buffered LDS in fragment
// order (cvt_pk on the fly).  B: prepacked bf16 fragments read straight
// from global (L2-broadcast), 2x global_load_b128 per fragment.
// EPI: 0 = (+bias), 1 = embed (+bias+pos+te), 2 = gelu(+bias), 3 = resid(+bias)
template<int EPI>
__global__ __launch_bounds__(256)
void gemm_k(const float* __restrict__ A, int lda,
            const unsigned short* __restrict__ Wp,
            const float* __restrict__ bias,
            const float* __restrict__ resid,
            const float* __restrict__ ax1,
            const float* __restrict__ ax2,
            const float* __restrict__ te,
            float* __restrict__ C, int ldc, int Kdim) {
  __shared__ __align__(32) unsigned short As[2][8][32][16];  // 16 KB
  const int tid  = threadIdx.x;
  const int lane = tid & 31;
  const int wid  = tid >> 5;
  const int wm   = wid & 3;        // 4 row groups of 32
  const int wn   = wid >> 2;       // 2 col groups of 32
  const int blockM = blockIdx.y * 128;
  const int blockN = blockIdx.x * 64;
  // A producer slot (one per thread): fills As[buf][a_ms][a_fl][0..15]
  const int a_ms = tid >> 5, a_fl = tid & 31;
  const int a_row = blockM + a_ms * 16 + (a_fl & 15);
  const int a_hi  = (a_fl >> 4) * 8;
  const float* arow = A + (size_t)a_row * lda;
  // prepacked B fragment pointers for this wave's two n-subtiles
  const int nkt = Kdim >> 5;
  const unsigned short* wp0 =
      Wp + ((size_t)((blockN >> 4) + wn * 2 + 0) * nkt * 32 + lane) * 16;
  const unsigned short* wp1 =
      Wp + ((size_t)((blockN >> 4) + wn * 2 + 1) * nkt * 32 + lane) * 16;

  float4 fa0, fa1, fa2, fa3;
  // prologue: fetch + stage A tile 0
  {
    const float* ap = arow + a_hi;
    fa0 = *(const float4*)(ap);
    fa1 = *(const float4*)(ap + 4);
    fa2 = *(const float4*)(ap + 16);
    fa3 = *(const float4*)(ap + 20);
  }
  *(uint4*)&As[0][a_ms][a_fl][0] =
      make_uint4(pk2(fa0.x,fa0.y), pk2(fa0.z,fa0.w), pk2(fa1.x,fa1.y), pk2(fa1.z,fa1.w));
  *(uint4*)&As[0][a_ms][a_fl][8] =
      make_uint4(pk2(fa2.x,fa2.y), pk2(fa2.z,fa2.w), pk2(fa3.x,fa3.y), pk2(fa3.z,fa3.w));

  v8f c00 = {}, c01 = {}, c10 = {}, c11 = {};
  for (int kt = 0; kt < nkt; ++kt) {
    __syncthreads();
    const int cur = kt & 1;
    if (kt + 2 < nkt) {              // warm caches for tile kt+2
      const int k2 = (kt + 2) << 5;
      __builtin_prefetch(arow + k2 + a_hi, 0, 3);
      __builtin_prefetch(wp0 + (size_t)(kt + 2) * 512, 0, 3);
      __builtin_prefetch(wp1 + (size_t)(kt + 2) * 512, 0, 3);
    }
    if (kt + 1 < nkt) {              // global fetch of A tile kt+1
      const int k1 = (kt + 1) << 5;
      const float* ap = arow + k1 + a_hi;
      fa0 = *(const float4*)(ap);
      fa1 = *(const float4*)(ap + 4);
      fa2 = *(const float4*)(ap + 16);
      fa3 = *(const float4*)(ap + 20);
    }
    // B fragments straight from global (prepacked bf16)
    v16bf b0 = *(const v16bf*)(wp0 + (size_t)kt * 512);
    v16bf b1 = *(const v16bf*)(wp1 + (size_t)kt * 512);
    // A fragments from LDS
    v16bf a0 = *(const v16bf*)&As[cur][wm*2+0][lane][0];
    v16bf a1 = *(const v16bf*)&As[cur][wm*2+1][lane][0];
    c00 = wmma_bf16(a0, b0, c00);
    c01 = wmma_bf16(a0, b1, c01);
    c10 = wmma_bf16(a1, b0, c10);
    c11 = wmma_bf16(a1, b1, c11);
    if (kt + 1 < nkt) {              // stage A tile kt+1 into the other buffer
      const int nxt = cur ^ 1;
      *(uint4*)&As[nxt][a_ms][a_fl][0] =
          make_uint4(pk2(fa0.x,fa0.y), pk2(fa0.z,fa0.w), pk2(fa1.x,fa1.y), pk2(fa1.z,fa1.w));
      *(uint4*)&As[nxt][a_ms][a_fl][8] =
          make_uint4(pk2(fa2.x,fa2.y), pk2(fa2.z,fa2.w), pk2(fa3.x,fa3.y), pk2(fa3.z,fa3.w));
    }
  }
  // epilogue: C frag element r @ lane L -> row = r + 8*(L>>4), col = L&15
  const int chi = (lane >> 4) * 8;
  const int cn  = lane & 15;
  v8f accs[4] = { c00, c01, c10, c11 };
  #pragma unroll
  for (int t = 0; t < 4; ++t) {
    const int ms  = wm * 2 + (t >> 1);
    const int ns  = wn * 2 + (t & 1);
    const int col = blockN + ns * 16 + cn;
    const float badd = (bias != nullptr) ? bias[col] : 0.f;
    #pragma unroll
    for (int r = 0; r < 8; ++r) {
      const int m = blockM + ms * 16 + chi + r;
      float vout = accs[t][r] + badd;
      if (EPI == 1) {
        const int s = m & (S_ - 1);
        vout += ax1[(s / W_) * D_ + col] + ax2[(s % W_) * D_ + col]
              + te[(m / S_) * D_ + col];
      } else if (EPI == 2) {
        vout = gelu_f(vout);
      } else if (EPI == 3) {
        vout += resid[(size_t)m * ldc + col];
      }
      C[(size_t)m * ldc + col] = vout;
    }
  }
}

// ================================================================ LayerNorm
__global__ __launch_bounds__(128)
void ln_k(const float* __restrict__ X, const float* __restrict__ g,
          const float* __restrict__ bta, float* __restrict__ Y) {
  const int row = blockIdx.x, t = threadIdx.x;
  const float* xp = X + (size_t)row * D_;
  float4 v = *(const float4*)(xp + t * 4);
  __shared__ float red[128];
  red[t] = v.x + v.y + v.z + v.w;
  __syncthreads();
  for (int o = 64; o > 0; o >>= 1) { if (t < o) red[t] += red[t + o]; __syncthreads(); }
  const float mu = red[0] * (1.f / D_);
  __syncthreads();
  const float dx = v.x - mu, dy = v.y - mu, dz = v.z - mu, dw = v.w - mu;
  red[t] = dx*dx + dy*dy + dz*dz + dw*dw;
  __syncthreads();
  for (int o = 64; o > 0; o >>= 1) { if (t < o) red[t] += red[t + o]; __syncthreads(); }
  const float rs = rsqrtf(red[0] * (1.f / D_) + 1e-5f);
  float4 gg = *(const float4*)(g + t * 4);
  float4 bb = *(const float4*)(bta + t * 4);
  float4 o4 = make_float4(dx*rs*gg.x + bb.x, dy*rs*gg.y + bb.y,
                          dz*rs*gg.z + bb.z, dw*rs*gg.w + bb.w);
  *(float4*)(Y + (size_t)row * D_ + t * 4) = o4;
}

// ================================================================ local attn
// scores[z,128,384] = q_win(128x64) . k_lookaround(384x64)^T  (WMMA batched)
__global__ __launch_bounds__(256)
void la_score_k(const float* __restrict__ Q, const float* __restrict__ Kb,
                float* __restrict__ scores) {
  __shared__ __align__(32) unsigned short As[8][32][16];
  __shared__ __align__(32) unsigned short Bs[4][32][16];
  const int z = blockIdx.z;                 // ((b*LH + h)*NW + wdx)
  const int wdx = z % NW_;
  const int bh  = z / NW_;
  const int h   = bh % LH_;
  const int b   = bh / LH_;
  const int blockN = blockIdx.x * 64;       // key tile within [0,384)
  const int tid = threadIdx.x, lane = tid & 31, wid = tid >> 5;
  const int wm = wid & 3, wn = wid >> 2;
  const int a_ms = tid >> 5, a_fl = tid & 31;
  const int a_r  = a_ms * 16 + (a_fl & 15);
  const int a_hi = (a_fl >> 4) * 8;
  const int b_slot = tid >> 1, b_half = tid & 1;
  const int b_ns = b_slot >> 5, b_fl = b_slot & 31;
  const int b_j  = blockN + b_ns * 16 + (b_fl & 15);
  const int b_kof = (b_fl >> 4) * 8 + b_half * 16;
  const int p = (wdx - 1) * W_ + b_j;
  const bool bval = (p >= 0) && (p < S_);
  const float* qbase = Q + (size_t)(b * S_ + wdx * W_ + a_r) * D_ + h * DH_;
  const float* kbase = Kb + (size_t)(b * S_ + (bval ? p : 0)) * D_ + h * DH_;

  v8f c00 = {}, c01 = {}, c10 = {}, c11 = {};
  for (int k0 = 0; k0 < DH_; k0 += 32) {
    __syncthreads();
    { const float* ap = qbase + k0 + a_hi;
      float4 f0 = *(const float4*)(ap);
      float4 f1 = *(const float4*)(ap + 4);
      float4 f2 = *(const float4*)(ap + 16);
      float4 f3 = *(const float4*)(ap + 20);
      *(uint4*)&As[a_ms][a_fl][0] =
          make_uint4(pk2(f0.x,f0.y), pk2(f0.z,f0.w), pk2(f1.x,f1.y), pk2(f1.z,f1.w));
      *(uint4*)&As[a_ms][a_fl][8] =
          make_uint4(pk2(f2.x,f2.y), pk2(f2.z,f2.w), pk2(f3.x,f3.y), pk2(f3.z,f3.w));
    }
    { float4 g0 = make_float4(0.f,0.f,0.f,0.f), g1 = g0;
      if (bval) {
        const float* bp = kbase + k0 + b_kof;
        g0 = *(const float4*)(bp);
        g1 = *(const float4*)(bp + 4);
      }
      *(uint4*)&Bs[b_ns][b_fl][b_half * 8] =
          make_uint4(pk2(g0.x,g0.y), pk2(g0.z,g0.w), pk2(g1.x,g1.y), pk2(g1.z,g1.w));
    }
    __syncthreads();
    v16bf a0 = *(const v16bf*)&As[wm*2+0][lane][0];
    v16bf a1 = *(const v16bf*)&As[wm*2+1][lane][0];
    v16bf b0 = *(const v16bf*)&Bs[wn*2+0][lane][0];
    v16bf b1 = *(const v16bf*)&Bs[wn*2+1][lane][0];
    c00 = wmma_bf16(a0, b0, c00);
    c01 = wmma_bf16(a0, b1, c01);
    c10 = wmma_bf16(a1, b0, c10);
    c11 = wmma_bf16(a1, b1, c11);
  }
  const int chi = (lane >> 4) * 8, cn = lane & 15;
  float* sb = scores + (size_t)z * (128 * 384);
  v8f accs[4] = { c00, c01, c10, c11 };
  #pragma unroll
  for (int t = 0; t < 4; ++t) {
    const int ms = wm * 2 + (t >> 1), ns = wn * 2 + (t & 1);
    const int col = blockN + ns * 16 + cn;
    #pragma unroll
    for (int r = 0; r < 8; ++r)
      sb[(size_t)(ms * 16 + chi + r) * 384 + col] = accs[t][r];
  }
}

// masked softmax over the 384 look-around keys (scale DH^-0.5)
__global__ __launch_bounds__(128)
void la_softmax_k(float* __restrict__ scores) {
  const int row = blockIdx.x;               // z*128 + query-row
  const int wdx = (row >> 7) % NW_;
  float* p = scores + (size_t)row * 384;
  const int t = threadIdx.x;
  __shared__ float red[128];
  float vals[3]; bool ok3[3];
  float mx = -3.0e38f;
  #pragma unroll
  for (int i = 0; i < 3; ++i) {
    const int j = t + i * 128;
    const int win = wdx + (j / W_) - 1;
    const bool ok = (win >= 0) && (win < NW_);
    const float v = ok ? p[j] * 0.125f : -3.0e38f;
    vals[i] = v; ok3[i] = ok;
    mx = fmaxf(mx, v);
  }
  red[t] = mx; __syncthreads();
  for (int o = 64; o > 0; o >>= 1) { if (t < o) red[t] = fmaxf(red[t], red[t+o]); __syncthreads(); }
  mx = red[0];
  __syncthreads();
  float e[3]; float sum = 0.f;
  #pragma unroll
  for (int i = 0; i < 3; ++i) { e[i] = ok3[i] ? expf(vals[i] - mx) : 0.f; sum += e[i]; }
  red[t] = sum; __syncthreads();
  for (int o = 64; o > 0; o >>= 1) { if (t < o) red[t] += red[t+o]; __syncthreads(); }
  const float inv = 1.f / red[0];
  #pragma unroll
  for (int i = 0; i < 3; ++i) p[t + i * 128] = e[i] * inv;
}

// out[z,128,64] = attn(128x384) . v_lookaround(384x64)   (WMMA batched)
__global__ __launch_bounds__(256)
void la_out_k(const float* __restrict__ attn, const float* __restrict__ Vb,
              float* __restrict__ aout) {
  __shared__ __align__(32) unsigned short As[8][32][16];
  __shared__ __align__(32) unsigned short Bs[4][32][16];
  const int z = blockIdx.z;
  const int wdx = z % NW_;
  const int bh  = z / NW_;
  const int h   = bh % LH_;
  const int b   = bh / LH_;
  const int tid = threadIdx.x, lane = tid & 31, wid = tid >> 5;
  const int wm = wid & 3, wn = wid >> 2;
  const int a_ms = tid >> 5, a_fl = tid & 31;
  const int a_r  = a_ms * 16 + (a_fl & 15);
  const int a_hi = (a_fl >> 4) * 8;
  const int b_slot = tid >> 1, b_half = tid & 1;
  const int b_ns = b_slot >> 5, b_fl = b_slot & 31;
  const int b_e  = b_ns * 16 + (b_fl & 15);
  const int b_kof = (b_fl >> 4) * 8 + b_half * 16;
  const float* abase = attn + (size_t)z * (128 * 384) + (size_t)a_r * 384;

  v8f c00 = {}, c01 = {}, c10 = {}, c11 = {};
  for (int k0 = 0; k0 < 384; k0 += 32) {
    __syncthreads();
    { const float* ap = abase + k0 + a_hi;
      float4 f0 = *(const float4*)(ap);
      float4 f1 = *(const float4*)(ap + 4);
      float4 f2 = *(const float4*)(ap + 16);
      float4 f3 = *(const float4*)(ap + 20);
      *(uint4*)&As[a_ms][a_fl][0] =
          make_uint4(pk2(f0.x,f0.y), pk2(f0.z,f0.w), pk2(f1.x,f1.y), pk2(f1.z,f1.w));
      *(uint4*)&As[a_ms][a_fl][8] =
          make_uint4(pk2(f2.x,f2.y), pk2(f2.z,f2.w), pk2(f3.x,f3.y), pk2(f3.z,f3.w));
    }
    { const int j0 = k0 + b_kof;                    // 8-run stays in one window
      const int p0 = (wdx - 1) * W_ + j0;
      const bool ok = (p0 >= 0) && (p0 < S_);
      float f[8];
      #pragma unroll
      for (int jj = 0; jj < 8; ++jj)
        f[jj] = ok ? Vb[(size_t)(b * S_ + p0 + jj) * D_ + h * DH_ + b_e] : 0.f;
      *(uint4*)&Bs[b_ns][b_fl][b_half * 8] =
          make_uint4(pk2(f[0],f[1]), pk2(f[2],f[3]), pk2(f[4],f[5]), pk2(f[6],f[7]));
    }
    __syncthreads();
    v16bf a0 = *(const v16bf*)&As[wm*2+0][lane][0];
    v16bf a1 = *(const v16bf*)&As[wm*2+1][lane][0];
    v16bf b0 = *(const v16bf*)&Bs[wn*2+0][lane][0];
    v16bf b1 = *(const v16bf*)&Bs[wn*2+1][lane][0];
    c00 = wmma_bf16(a0, b0, c00);
    c01 = wmma_bf16(a0, b1, c01);
    c10 = wmma_bf16(a1, b0, c10);
    c11 = wmma_bf16(a1, b1, c11);
  }
  const int chi = (lane >> 4) * 8, cn = lane & 15;
  v8f accs[4] = { c00, c01, c10, c11 };
  #pragma unroll
  for (int t = 0; t < 4; ++t) {
    const int ms = wm * 2 + (t >> 1), ns = wn * 2 + (t & 1);
    const int col = ns * 16 + cn;
    #pragma unroll
    for (int r = 0; r < 8; ++r) {
      const int m = ms * 16 + chi + r;
      aout[(size_t)(b * S_ + wdx * W_ + m) * D_ + h * DH_ + col] = accs[t][r];
    }
  }
}

// ================================================================ linear attn
// per (b, global-head): column max & 1/sum of exp over the sequence
__global__ __launch_bounds__(256)
void lin_stats_k(const float* __restrict__ Kb, float* __restrict__ cmax,
                 float* __restrict__ cis) {
  const int z = blockIdx.x;                  // b*4 + hh
  const int b = z >> 2, h = (z & 3) + LH_;
  const int t = threadIdx.x, d = t & 63, part = t >> 6;
  const float sc = 0.35355339059327376f;     // DH^-0.25
  __shared__ float red[256];
  const float* kp = Kb + (size_t)b * S_ * D_ + h * DH_ + d;
  float mx = -3.0e38f;
  for (int n = part * 512; n < part * 512 + 512; ++n)
    mx = fmaxf(mx, kp[(size_t)n * D_]);
  red[t] = mx; __syncthreads();
  if (part == 0)
    red[d] = fmaxf(fmaxf(red[d], red[d+64]), fmaxf(red[d+128], red[d+192]));
  __syncthreads();
  mx = red[d] * sc;                           // sc > 0 -> max commutes
  __syncthreads();
  float sum = 0.f;
  for (int n = part * 512; n < part * 512 + 512; ++n)
    sum += expf(kp[(size_t)n * D_] * sc - mx);
  red[t] = sum; __syncthreads();
  if (part == 0) {
    const float s = red[d] + red[d+64] + red[d+128] + red[d+192];
    cmax[z * 64 + d] = mx;
    cis[z * 64 + d]  = 1.f / s;
  }
}

// ctx[z,64,64] = softmax_seq(k)^T . v   (M=64,N=64,K=2048 WMMA, batched)
__global__ __launch_bounds__(256)
void lin_ctx_k(const float* __restrict__ Kb, const float* __restrict__ Vb,
               const float* __restrict__ cmax, const float* __restrict__ cis,
               float* __restrict__ ctx) {
  __shared__ __align__(32) unsigned short As[4][32][16];
  __shared__ __align__(32) unsigned short Bs[4][32][16];
  const int z = blockIdx.x, b = z >> 2, h = (z & 3) + LH_;
  const int tid = threadIdx.x, lane = tid & 31, wid = tid >> 5;
  const int wm = wid & 3, wn = wid >> 2;       // wm: m-subtile, wn: col half
  const float sc = 0.35355339059327376f;
  const int sl_ms = (tid & 127) >> 5, sl_fl = tid & 31;
  const int sl_d  = sl_ms * 16 + (sl_fl & 15);     // A: row d / B: col e
  const int a_hi  = (sl_fl >> 4) * 8;
  const int b_kof = (sl_fl >> 4) * 16;
  const float am = cmax[z * 64 + sl_d];
  const float ai = cis[z * 64 + sl_d];

  v8f c0 = {}, c1 = {};
  for (int k0 = 0; k0 < S_; k0 += 32) {
    __syncthreads();
    if (tid < 128) {               // A: softk[d][n] computed on the fly
      float f[16];
      #pragma unroll
      for (int i = 0; i < 16; ++i) {
        const int n = k0 + a_hi + (i & 7) + ((i >> 3) * 16);
        const float kv = Kb[(size_t)(b * S_ + n) * D_ + h * DH_ + sl_d];
        f[i] = expf(kv * sc - am) * ai;
      }
      *(uint4*)&As[sl_ms][sl_fl][0] =
          make_uint4(pk2(f[0],f[1]), pk2(f[2],f[3]), pk2(f[4],f[5]), pk2(f[6],f[7]));
      *(uint4*)&As[sl_ms][sl_fl][8] =
          make_uint4(pk2(f[8],f[9]), pk2(f[10],f[11]), pk2(f[12],f[13]), pk2(f[14],f[15]));
    } else {                       // B: v[n][e]
      float f[16];
      #pragma unroll
      for (int i = 0; i < 16; ++i) {
        const int n = k0 + a_hi + (i & 7) + ((i >> 3) * 16);
        f[i] = Vb[(size_t)(b * S_ + n) * D_ + h * DH_ + sl_d];
      }
      *(uint4*)&Bs[sl_ms][sl_fl][0] =
          make_uint4(pk2(f[0],f[1]), pk2(f[2],f[3]), pk2(f[4],f[5]), pk2(f[6],f[7]));
      *(uint4*)&Bs[sl_ms][sl_fl][8] =
          make_uint4(pk2(f[8],f[9]), pk2(f[10],f[11]), pk2(f[12],f[13]), pk2(f[14],f[15]));
    }
    __syncthreads();
    v16bf a  = *(const v16bf*)&As[wm][lane][0];
    v16bf b0 = *(const v16bf*)&Bs[wn*2+0][lane][0];
    v16bf b1 = *(const v16bf*)&Bs[wn*2+1][lane][0];
    c0 = wmma_bf16(a, b0, c0);
    c1 = wmma_bf16(a, b1, c1);
  }
  const int chi = (lane >> 4) * 8, cn = lane & 15;
  float* cp = ctx + (size_t)z * 4096;
  v8f accs[2] = { c0, c1 };
  #pragma unroll
  for (int t = 0; t < 2; ++t) {
    const int col = (wn * 2 + t) * 16 + cn;
    #pragma unroll
    for (int r = 0; r < 8; ++r)
      cp[(wm * 16 + chi + r) * 64 + col] = accs[t][r];
  }
}

// out[b,n,h,:] = softmax_feat(q) . ctx    (wave per token, wave32 shfl)
__global__ __launch_bounds__(256)
void lin_apply_k(const float* __restrict__ Q, const float* __restrict__ ctx,
                 float* __restrict__ aout) {
  __shared__ float sq[8][64];
  const int tid = threadIdx.x, lane = tid & 31, wid = tid >> 5;
  const int g = blockIdx.x * 8 + wid;         // token over (z, n)
  const int z = g >> 11, n = g & 2047;
  const int b = z >> 2, h = (z & 3) + LH_;
  const float sc = 0.35355339059327376f;
  const float* qp = Q + (size_t)(b * S_ + n) * D_ + h * DH_;
  float q0 = qp[lane] * sc, q1 = qp[lane + 32] * sc;
  float m = fmaxf(q0, q1);
  for (int o = 16; o > 0; o >>= 1) m = fmaxf(m, __shfl_xor(m, o, 32));
  const float e0 = expf(q0 - m), e1 = expf(q1 - m);
  float s = e0 + e1;
  for (int o = 16; o > 0; o >>= 1) s += __shfl_xor(s, o, 32);
  const float inv = 1.f / s;
  sq[wid][lane] = e0 * inv; sq[wid][lane + 32] = e1 * inv;
  __syncthreads();
  const float* cp = ctx + (size_t)z * 4096;
  float acc0 = 0.f, acc1 = 0.f;
  for (int d = 0; d < 64; ++d) {
    const float f = sq[wid][d];
    acc0 += f * cp[d * 64 + lane];
    acc1 += f * cp[d * 64 + lane + 32];
  }
  float* op = aout + (size_t)(b * S_ + n) * D_ + h * DH_;
  op[lane] = acc0; op[lane + 32] = acc1;
}

// ================================================================ time MLP
__global__ __launch_bounds__(256)
void te_sin_k(const int* __restrict__ t, float* __restrict__ sinb) {
  const int b = blockIdx.x, j = threadIdx.x;   // j < 256
  const float tt = (float)t[b] * 4.0f;          // t/1000*4000
  const float fr = expf((float)j * (-logf(10000.f) / 255.f));
  const float e = tt * fr;
  sinb[b * D_ + j] = sinf(e);
  sinb[b * D_ + 256 + j] = cosf(e);
}
__global__ __launch_bounds__(256)
void te_mid_k(const float* __restrict__ sinb, const float* __restrict__ w1,
              const float* __restrict__ b1, float* __restrict__ tmid) {
  const int o = blockIdx.x * 256 + threadIdx.x;  // 8*2048
  const int b = o >> 11, c = o & 2047;
  float acc = b1[c];
  for (int i = 0; i < D_; ++i) acc += sinb[b * D_ + i] * w1[(size_t)i * FD_ + c];
  const float sp = (acc > 20.f) ? acc : log1pf(expf(acc));
  tmid[o] = acc * tanhf(sp);                     // mish
}
__global__ __launch_bounds__(256)
void te_out_k(const float* __restrict__ tmid, const float* __restrict__ w2,
              const float* __restrict__ b2, float* __restrict__ teB) {
  const int o = blockIdx.x * 256 + threadIdx.x;  // 8*512
  const int b = o >> 9, c = o & 511;
  float acc = b2[c];
  for (int i = 0; i < FD_; ++i) acc += tmid[b * FD_ + i] * w2[(size_t)i * D_ + c];
  teB[o] = acc;
}

// ================================================================ final
__global__ __launch_bounds__(256)
void trans_k(const float* __restrict__ o, const float* __restrict__ alpha,
             float* __restrict__ out) {
  const int idx = blockIdx.x * 256 + threadIdx.x;   // over B*K*S
  const int s = idx & (S_ - 1);
  const int rest = idx >> 11;
  const int kk = rest & (K_ - 1);
  const int b = rest >> 7;
  out[idx] = alpha[0] * o[(size_t)(b * S_ + s) * K_ + kk];
}

// ================================================================ launcher
extern "C" void kernel_launch(void* const* d_in, const int* in_sizes, int n_in,
                              void* d_out, int out_size, void* d_ws, size_t ws_size,
                              hipStream_t stream) {
  (void)in_sizes; (void)n_in; (void)out_size; (void)ws_size;
  const float* x        = (const float*)d_in[0];
  const int*   tsteps   = (const int*)d_in[1];
  const float* token_w  = (const float*)d_in[2];
  const float* token_b  = (const float*)d_in[3];
  const float* ax1      = (const float*)d_in[4];
  const float* ax2      = (const float*)d_in[5];
  const float* time_w1  = (const float*)d_in[6];
  const float* time_b1  = (const float*)d_in[7];
  const float* time_w2  = (const float*)d_in[8];
  const float* time_b2  = (const float*)d_in[9];
  const float* ln1_g    = (const float*)d_in[10];
  const float* ln1_b    = (const float*)d_in[11];
  const float* wq       = (const float*)d_in[12];
  const float* wk       = (const float*)d_in[13];
  const float* wv       = (const float*)d_in[14];
  const float* wo       = (const float*)d_in[15];
  const float* wo_b     = (const float*)d_in[16];
  const float* ln2_g    = (const float*)d_in[17];
  const float* ln2_b    = (const float*)d_in[18];
  const float* ff_w1    = (const float*)d_in[19];
  const float* ff_b1    = (const float*)d_in[20];
  const float* ff_w2    = (const float*)d_in[21];
  const float* ff_b2    = (const float*)d_in[22];
  const float* norm_g   = (const float*)d_in[23];
  const float* norm_b   = (const float*)d_in[24];
  const float* out_w    = (const float*)d_in[25];
  const float* out_b    = (const float*)d_in[26];
  const float* alpha    = (const float*)d_in[27];

  float* ws = (float*)d_ws;
  size_t off = 0;
  auto alloc = [&](size_t nf) { float* p = ws + off; off += nf; return p; };
  float* hb   = alloc((size_t)BS_ * D_);
  float* nb   = alloc((size_t)BS_ * D_);
  float* qb   = alloc((size_t)BS_ * D_);
  float* kb   = alloc((size_t)BS_ * D_);
  float* vb   = alloc((size_t)BS_ * D_);
  float* aob  = alloc((size_t)BS_ * D_);
  float* big  = alloc((size_t)BS_ * FD_);       // FFN mid; also attn scores
  float* sinb = alloc((size_t)B_ * D_);
  float* tmid = alloc((size_t)B_ * FD_);
  float* teB  = alloc((size_t)B_ * D_);
  float* cmax = alloc(32 * 64);
  float* cis  = alloc(32 * 64);
  float* ctx  = alloc(32 * 64 * 64);
  float* obsk = alloc((size_t)BS_ * K_);
  // prepacked bf16 weights (2 bytes/elem, sizes are multiples of 8 floats)
  unsigned short* wp_tok = (unsigned short*)alloc((size_t)K_ * D_ / 2);
  unsigned short* wp_q   = (unsigned short*)alloc((size_t)L_ * D_ * D_ / 2);
  unsigned short* wp_k   = (unsigned short*)alloc((size_t)L_ * D_ * D_ / 2);
  unsigned short* wp_v   = (unsigned short*)alloc((size_t)L_ * D_ * D_ / 2);
  unsigned short* wp_o   = (unsigned short*)alloc((size_t)L_ * D_ * D_ / 2);
  unsigned short* wp_f1  = (unsigned short*)alloc((size_t)L_ * D_ * FD_ / 2);
  unsigned short* wp_f2  = (unsigned short*)alloc((size_t)L_ * FD_ * D_ / 2);
  unsigned short* wp_out = (unsigned short*)alloc((size_t)D_ * K_ / 2);

  // ---- weight prepack (once per call; reused by 128 M-blocks each) ----
  prep_w_k<<<(K_ * D_ / 16) / 256, 256, 0, stream>>>(token_w, D_, K_, wp_tok);
  for (int l = 0; l < L_; ++l) {
    const size_t wdd = (size_t)l * D_ * D_;
    prep_w_k<<<(D_ * D_ / 16) / 256, 256, 0, stream>>>(wq + wdd, D_, D_, wp_q + wdd);
    prep_w_k<<<(D_ * D_ / 16) / 256, 256, 0, stream>>>(wk + wdd, D_, D_, wp_k + wdd);
    prep_w_k<<<(D_ * D_ / 16) / 256, 256, 0, stream>>>(wv + wdd, D_, D_, wp_v + wdd);
    prep_w_k<<<(D_ * D_ / 16) / 256, 256, 0, stream>>>(wo + wdd, D_, D_, wp_o + wdd);
    const size_t wdf = (size_t)l * D_ * FD_;
    prep_w_k<<<(D_ * FD_ / 16) / 256, 256, 0, stream>>>(ff_w1 + wdf, FD_, D_, wp_f1 + wdf);
    prep_w_k<<<(FD_ * D_ / 16) / 256, 256, 0, stream>>>(ff_w2 + wdf, D_, FD_, wp_f2 + wdf);
  }
  prep_w_k<<<(D_ * K_ / 16) / 256, 256, 0, stream>>>(out_w, K_, D_, wp_out);

  // ---- time embedding (tiny) ----
  te_sin_k<<<B_, 256, 0, stream>>>(tsteps, sinb);
  te_mid_k<<<(B_ * FD_) / 256, 256, 0, stream>>>(sinb, time_w1, time_b1, tmid);
  te_out_k<<<(B_ * D_) / 256, 256, 0, stream>>>(tmid, time_w2, time_b2, teB);

  // h = x @ token_w + token_b + pos + te
  gemm_k<1><<<dim3(D_ / 64, BS_ / 128), 256, 0, stream>>>(
      x, K_, wp_tok, token_b, nullptr, ax1, ax2, teB, hb, D_, K_);

  for (int l = 0; l < L_; ++l) {
    const size_t wdd = (size_t)l * D_ * D_;
    const size_t wdf = (size_t)l * D_ * FD_;
    ln_k<<<BS_, 128, 0, stream>>>(hb, ln1_g + l * D_, ln1_b + l * D_, nb);
    gemm_k<0><<<dim3(D_ / 64, BS_ / 128), 256, 0, stream>>>(
        nb, D_, wp_q + wdd, nullptr, nullptr, nullptr, nullptr, nullptr, qb, D_, D_);
    gemm_k<0><<<dim3(D_ / 64, BS_ / 128), 256, 0, stream>>>(
        nb, D_, wp_k + wdd, nullptr, nullptr, nullptr, nullptr, nullptr, kb, D_, D_);
    gemm_k<0><<<dim3(D_ / 64, BS_ / 128), 256, 0, stream>>>(
        nb, D_, wp_v + wdd, nullptr, nullptr, nullptr, nullptr, nullptr, vb, D_, D_);

    // local attention heads [0,4)
    la_score_k<<<dim3(6, 1, B_ * LH_ * NW_), 256, 0, stream>>>(qb, kb, big);
    la_softmax_k<<<B_ * LH_ * NW_ * 128, 128, 0, stream>>>(big);
    la_out_k<<<dim3(1, 1, B_ * LH_ * NW_), 256, 0, stream>>>(big, vb, aob);

    // linear attention heads [4,8)
    lin_stats_k<<<32, 256, 0, stream>>>(kb, cmax, cis);
    lin_ctx_k<<<32, 256, 0, stream>>>(kb, vb, cmax, cis, ctx);
    lin_apply_k<<<(32 * S_) / 8, 256, 0, stream>>>(qb, ctx, aob);

    // h += aout @ wo + wo_b
    gemm_k<3><<<dim3(D_ / 64, BS_ / 128), 256, 0, stream>>>(
        aob, D_, wp_o + wdd, wo_b + l * D_, hb, nullptr, nullptr, nullptr,
        hb, D_, D_);

    // FFN
    ln_k<<<BS_, 128, 0, stream>>>(hb, ln2_g + l * D_, ln2_b + l * D_, nb);
    gemm_k<2><<<dim3(FD_ / 64, BS_ / 128), 256, 0, stream>>>(
        nb, D_, wp_f1 + wdf, ff_b1 + l * FD_, nullptr, nullptr, nullptr, nullptr,
        big, FD_, D_);
    gemm_k<3><<<dim3(D_ / 64, BS_ / 128), 256, 0, stream>>>(
        big, FD_, wp_f2 + wdf, ff_b2 + l * D_, hb, nullptr, nullptr, nullptr,
        hb, D_, FD_);
  }

  ln_k<<<BS_, 128, 0, stream>>>(hb, norm_g, norm_b, nb);
  gemm_k<0><<<dim3(K_ / 64, BS_ / 128), 256, 0, stream>>>(
      nb, D_, wp_out, out_b, nullptr, nullptr, nullptr, nullptr, obsk, K_, D_);
  trans_k<<<(B_ * K_ * S_) / 256, 256, 0, stream>>>(obsk, alpha, (float*)d_out);
}
